// HeunIntegrator_40080634806918
// MI455X (gfx1250) — matrix operand order repaired and checked by reading, not compiled, not run on hardware
//
#include <hip/hip_runtime.h>
#include <hip/hip_bf16.h>
#include <cstdint>

// ---------------------------------------------------------------------------
// Problem constants (match reference)
// ---------------------------------------------------------------------------
#define B_ROWS 16384
#define DDIM   256
#define HDIM   1024
#define DTF    0.01f
#define PLASTICITY 0.1f
#define PI_F         3.14159265358979323846f
#define TWO_PI_F     6.283185307179586f
#define INV_TWO_PI_F 0.15915494309189535f

typedef __attribute__((ext_vector_type(16))) __bf16        v16bf;
typedef __attribute__((ext_vector_type(8)))  float         v8f;
typedef __attribute__((ext_vector_type(8)))  unsigned int  v8u;
typedef __attribute__((ext_vector_type(4)))  unsigned int  v4u;
typedef __attribute__((ext_vector_type(4)))  unsigned int  u32x4;
typedef __attribute__((ext_vector_type(8)))  int           i32x8;
typedef __attribute__((ext_vector_type(4)))  int           i32x4;

// ---------------------------------------------------------------------------
// Packed bf16 weights in workspace, ordered as TDM stream chunks (dwords).
//
// Phase-1 chunk hk (32 tiles = 8192 dw = 32 KB), hk in [0,32):
//   tiles  0..7  : U  B-tiles, nc = 2*hk,   kc = 0..7
//   tiles  8..15 : U  B-tiles, nc = 2*hk+1, kc = 0..7
//   tiles 16..31 : W  B-tiles, nc = 0..15,  kc = hk
// Phase-2 chunk nc (16 tiles = 4096 dw = 16 KB), nc in [0,16):
//   tiles 0..7  : Wf^T B-tiles, kc = 0..7
//   tiles 8..15 : Wi^T B-tiles, kc = 0..7
//
// Within a 32x16 B tile: dword = (tile_local*256 + lane*8 + g);
// element e=2g+j -> K = kc*32 + 8*(lane>>4) + e + (e>=8 ? 8 : 0), N = nc*16 + (lane&15).
// ---------------------------------------------------------------------------
#define P1_CHUNK_DW 8192u
#define P2_CHUNK_DW 4096u
#define P2_OFF      262144u            // 32 * 8192
#define TOTAL_PK    327680u            // + 16 * 4096  (= 1.25 MB)

// LDS float-offset layout (per block, 4 waves):
//   state: wave w at w*12288: cx=+0, cv=+4096, vp=+8192        (192 KB)
//   hst  : 49152 + w*256   (512 bf16 tanh staging per wave)     (  4 KB)
//   vsq  : 50176 + w*16                                         ( 256 B)
//   wbuf : 50240 (chunk buf 0), 58432 (chunk buf 1)             ( 64 KB)
#define WAVE_STATE_F 4096
#define HST_OFF   49152
#define VSQ_OFF   50176
#define WBUF0_OFF 50240
#define WBUF1_OFF (WBUF0_OFF + 8192)
#define LDS_TOTAL_F (WBUF1_OFF + 8192)   // 66624 floats = 266496 B

// ---------------------------------------------------------------------------
// Weight prep: f32 -> bf16, swizzled into the TDM stream layout above
// ---------------------------------------------------------------------------
__device__ __forceinline__ unsigned short f2bf_bits(float f) {
  __bf16 b = (__bf16)f;
  return __builtin_bit_cast(unsigned short, b);
}

__global__ void prep_weights_kernel(const float* __restrict__ U,
                                    const float* __restrict__ W,
                                    const float* __restrict__ Wf,
                                    const float* __restrict__ Wi,
                                    unsigned int* __restrict__ wpk) {
  unsigned int t = blockIdx.x * blockDim.x + threadIdx.x;
  if (t >= TOTAL_PK) return;
  unsigned int idx, kc, nc;
  int mat;
  if (t < P2_OFF) {
    unsigned int chunk  = t >> 13;       // / 8192
    unsigned int within = t & 8191u;
    unsigned int tile   = within >> 8;   // / 256
    idx = within & 255u;
    if (tile < 16u) { mat = 0; nc = 2u * chunk + (tile >> 3); kc = tile & 7u; }
    else            { mat = 1; nc = tile - 16u;               kc = chunk;     }
  } else {
    unsigned int r      = t - P2_OFF;
    unsigned int chunk  = r >> 12;       // / 4096
    unsigned int within = r & 4095u;
    unsigned int tile   = within >> 8;
    idx = within & 255u;
    if (tile < 8u) { mat = 2; nc = chunk; kc = tile;      }
    else           { mat = 3; nc = chunk; kc = tile - 8u; }
  }
  unsigned int g    = idx & 7u;
  unsigned int lane = idx >> 3;
  unsigned int half = lane >> 4;
  unsigned int n    = nc * 16u + (lane & 15u);
  unsigned int pack = 0u;
#pragma unroll
  for (unsigned int j = 0; j < 2; ++j) {
    unsigned int e = 2u * g + j;
    unsigned int k = kc * 32u + 8u * half + e + (e >= 8u ? 8u : 0u);
    float val;
    if (mat == 0)      val = U[k * HDIM + n];    // U: [D,H] row-major
    else if (mat == 1) val = W[k * DDIM + n];    // W: [H,D] row-major
    else if (mat == 2) val = Wf[n * DDIM + k];   // B = Wf^T
    else               val = Wi[n * DDIM + k];   // B = Wi^T
    pack |= ((unsigned int)f2bf_bits(val)) << (16u * j);
  }
  wpk[t] = pack;
}

// ---------------------------------------------------------------------------
// TDM: async 1-row tile copy global -> LDS (ndwords contiguous dwords)
// ---------------------------------------------------------------------------
__device__ __forceinline__ void tdm_load_chunk(const unsigned int* gsrc,
                                               unsigned int* ldst,
                                               unsigned int ndwords) {
  unsigned long long ga = (unsigned long long)(uintptr_t)gsrc;
  unsigned int lofs = (unsigned int)(uintptr_t)ldst;  // LDS byte offset (addr[31:0])
  u32x4 g0;
  g0[0] = 1u;                                             // count=1 (valid user D#)
  g0[1] = lofs;                                           // lds_addr
  g0[2] = (unsigned int)(ga & 0xFFFFFFFFull);             // global_addr[31:0]
  g0[3] = (unsigned int)((ga >> 32) & 0x01FFFFFFull)      // global_addr[56:32]
          | (2u << 30);                                   // type = 2 ("image")
  i32x8 g1;
  g1[0] = (int)(2u << 16);                                // data_size = 4 bytes
  g1[1] = (int)((ndwords & 0xFFFFu) << 16);               // tensor_dim0[15:0]
  g1[2] = (int)((ndwords >> 16) | (1u << 16));            // tensor_dim0[31:16] | tensor_dim1=1
  g1[3] = (int)((ndwords & 0xFFFFu) << 16);               // tile_dim0
  g1[4] = 0;                                              // tile_dim1=0, tile_dim2=0
  g1[5] = (int)ndwords;                                   // tensor_dim0_stride[31:0]
  g1[6] = 0;
  g1[7] = 0;
  i32x4 gz = {0, 0, 0, 0};
#if defined(__clang_major__) && (__clang_major__ >= 23)
  i32x8 gz8 = {0, 0, 0, 0, 0, 0, 0, 0};
  __builtin_amdgcn_tensor_load_to_lds(g0, g1, gz, gz, gz8, 0);
#else
  __builtin_amdgcn_tensor_load_to_lds(g0, g1, gz, gz, 0);
#endif
}

// ---------------------------------------------------------------------------
// WMMA helpers
// ---------------------------------------------------------------------------
__device__ __forceinline__ v8f vzero8() {
  v8f z;
#pragma unroll
  for (int i = 0; i < 8; ++i) z[i] = 0.0f;
  return z;
}

__device__ __forceinline__ v8f wmma_bf16(v16bf a, v16bf b, v8f c) {
  return __builtin_amdgcn_wmma_f32_16x16x32_bf16(false, a, false, b,
                                                 (short)0, c, false, false);
}

// B tile from LDS chunk buffer (ds_load_b128 x2 per lane)
__device__ __forceinline__ v16bf load_b_tile_lds(const unsigned int* chunk,
                                                 int tile, int lane) {
  const v8u* p = (const v8u*)(chunk + (size_t)tile * 256u + (size_t)lane * 8u);
  return __builtin_bit_cast(v16bf, *p);
}

// Build 16x32 bf16 A tile from f32 row-major [16 x 256] LDS buffer.
// A layout: lane m=lane&15; e<8 -> K = kc*32 + 8*half + e
//           e>=8 -> K = kc*32 + 16 + 8*half + (e-8)
__device__ __forceinline__ v16bf make_a_tile(const float* buf, int m, int half, int kc) {
  const float* p = buf + m * DDIM + kc * 32 + 8 * half;
  float4 q0 = *(const float4*)(p);
  float4 q1 = *(const float4*)(p + 4);
  float4 q2 = *(const float4*)(p + 16);
  float4 q3 = *(const float4*)(p + 20);
  v16bf a;
  a[0] = (__bf16)q0.x;  a[1] = (__bf16)q0.y;  a[2] = (__bf16)q0.z;  a[3] = (__bf16)q0.w;
  a[4] = (__bf16)q1.x;  a[5] = (__bf16)q1.y;  a[6] = (__bf16)q1.z;  a[7] = (__bf16)q1.w;
  a[8] = (__bf16)q2.x;  a[9] = (__bf16)q2.y;  a[10] = (__bf16)q2.z; a[11] = (__bf16)q2.w;
  a[12] = (__bf16)q3.x; a[13] = (__bf16)q3.y; a[14] = (__bf16)q3.z; a[15] = (__bf16)q3.w;
  return a;
}

__device__ __forceinline__ float wrapf(float x) {
  float t = x + PI_F;
  t -= floorf(t * INV_TWO_PI_F) * TWO_PI_F;
  return t - PI_F;
}
__device__ __forceinline__ float sigmoidf_(float x) {
  return 1.0f / (1.0f + __expf(-x));
}

// ---------------------------------------------------------------------------
// One dynamics evaluation + Heun partial update for a 16-row wave tile.
// State buffers: cx (doubles as x_pred between evals), cv, vp.
// MODE 0 (k1): vp = cv + dt*dv1 ; cv += 0.5dt*dv1 ; cx = wrap(cx + dt*cv_old)
// MODE 1 (k2): cv += 0.5dt*dv2 ; cx = wrap(x_pred + dt*(v_pred - cv_half))
// ---------------------------------------------------------------------------
template <int MODE>
__device__ __forceinline__ void eval_dynamics(
    const float* xb, const float* vb,
    float* cx, float* cv, float* vp,
    float* vsqp, __bf16* hst,
    float* lds, const unsigned int* __restrict__ wpk,
    const float* __restrict__ force_row,
    const float* __restrict__ bforget,
    int lane, int tid) {
  const int m15  = lane & 15;
  const int half = lane >> 4;
  unsigned int* wb0 = (unsigned int*)(lds + WBUF0_OFF);
  unsigned int* wb1 = (unsigned int*)(lds + WBUF1_OFF);

  // kick off phase-1 chunk 0 while we do elementwise prep
  if (tid < 32) tdm_load_chunk(wpk, wb0, P1_CHUNK_DW);

  // ---- per-row |v|^2 (lanes 0..15 each reduce one row) ----
  if (lane < 16) {
    const float4* vr = (const float4*)(vb + lane * DDIM);
    float s = 0.0f;
#pragma unroll
    for (int i = 0; i < DDIM / 4; ++i) {
      float4 q = vr[i];
      s += q.x * q.x + q.y * q.y + q.z * q.z + q.w * q.w;
    }
    vsqp[lane] = s;
  }
  __asm__ volatile("s_wait_dscnt 0x0" ::: "memory");

  // ---- A tiles for x (shared by x@U and x@Wf^T) ----
  v16bf a_x[8];
#pragma unroll
  for (int kc = 0; kc < 8; ++kc) a_x[kc] = make_a_tile(xb, m15, half, kc);

  v8f gamma[16];
#pragma unroll
  for (int i = 0; i < 16; ++i) gamma[i] = vzero8();

  if (tid < 32) __builtin_amdgcn_s_wait_tensorcnt(0);
  __syncthreads();

  // ---- phase 1: gamma = tanh(x@U) @ W, TDM-streamed weights ----
  for (int hk = 0; hk < HDIM / 32; ++hk) {
    const unsigned int* cur = (hk & 1) ? wb1 : wb0;
    if (tid < 32 && hk + 1 < HDIM / 32)
      tdm_load_chunk(wpk + (size_t)(hk + 1) * P1_CHUNK_DW,
                     (hk & 1) ? wb0 : wb1, P1_CHUNK_DW);

#pragma unroll
    for (int t2 = 0; t2 < 2; ++t2) {
      v8f acc = vzero8();
#pragma unroll
      for (int kc = 0; kc < 8; ++kc) {
        v16bf bt = load_b_tile_lds(cur, t2 * 8 + kc, lane);
        acc = wmma_bf16(a_x[kc], bt, acc);
      }
#pragma unroll
      for (int r = 0; r < 8; ++r) {
        int row = r + 8 * half;
        hst[row * 32 + t2 * 16 + m15] = (__bf16)tanhf(acc[r]);
      }
    }
    __asm__ volatile("s_wait_dscnt 0x0" ::: "memory");

    // staged tanh chunk as A tile (row m, local K in [0,32))
    const __bf16* hp = hst + m15 * 32 + 8 * half;
    v4u lo = *(const v4u*)(hp);
    v4u hi = *(const v4u*)(hp + 16);
    v8u pk;
#pragma unroll
    for (int i = 0; i < 4; ++i) { pk[i] = lo[i]; pk[4 + i] = hi[i]; }
    v16bf a_h = __builtin_bit_cast(v16bf, pk);

#pragma unroll
    for (int nc = 0; nc < 16; ++nc) {
      v16bf bt = load_b_tile_lds(cur, 16 + nc, lane);
      gamma[nc] = wmma_bf16(a_h, bt, gamma[nc]);
    }

    if (tid < 32 && hk + 1 < HDIM / 32) __builtin_amdgcn_s_wait_tensorcnt(0);
    __syncthreads();
  }

  // ---- A tiles for v (i_gate) ----
  v16bf a_v[8];
#pragma unroll
  for (int kc = 0; kc < 8; ++kc) a_v[kc] = make_a_tile(vb, m15, half, kc);

  // prologue for phase-2 stream
  if (tid < 32) {
    tdm_load_chunk(wpk + P2_OFF, wb0, P2_CHUNK_DW);
    __builtin_amdgcn_s_wait_tensorcnt(0);
  }
  __syncthreads();

  // ---- phase 2: gates + combine + Heun partial update ----
#pragma unroll
  for (int nc = 0; nc < 16; ++nc) {
    const unsigned int* cur = (nc & 1) ? wb1 : wb0;
    if (tid < 32 && nc + 1 < 16)
      tdm_load_chunk(wpk + P2_OFF + (size_t)(nc + 1) * P2_CHUNK_DW,
                     (nc & 1) ? wb0 : wb1, P2_CHUNK_DW);

    v8f facc = vzero8(), iacc = vzero8();
#pragma unroll
    for (int kc = 0; kc < 8; ++kc) {
      facc = wmma_bf16(a_x[kc], load_b_tile_lds(cur, kc, lane), facc);
      iacc = wmma_bf16(a_v[kc], load_b_tile_lds(cur, 8 + kc, lane), iacc);
    }
    const int n = nc * 16 + m15;
    const float bfg = bforget[n];
#pragma unroll
    for (int r = 0; r < 8; ++r) {
      const int row = r + 8 * half;
      const int off = row * DDIM + n;
      float vsq = vsqp[row];
      float amp = vsq > 1.0f ? 2.0f : 1.0f;  // |v| > 1  <=>  v^2 > 1
      float f_g = sigmoidf_(facc[r] + bfg);
      float i_g = sigmoidf_(iacc[r]);
      float vv  = vb[off];
      float c   = gamma[nc][r] * vsq * f_g + PLASTICITY * i_g * vv;
      float dv  = -(c * amp) + force_row[off];
      if (MODE == 0) {
        float cvo = cv[off];               // cv_old (== vv here)
        float cxo = cx[off];
        vp[off] = cvo + DTF * dv;          // v_pred
        cv[off] = cvo + 0.5f * DTF * dv;   // cv_half
        cx[off] = wrapf(cxo + DTF * cvo);  // x_pred (overwrites cx)
      } else {
        float cvh = cv[off];               // cv_half
        float xpv = cx[off];               // x_pred
        cv[off] = cvh + 0.5f * DTF * dv;   // final v
        cx[off] = wrapf(xpv + DTF * (vv - cvh));  // final x (vv == v_pred)
      }
    }
    if (tid < 32 && nc + 1 < 16) __builtin_amdgcn_s_wait_tensorcnt(0);
    __syncthreads();
  }
}

// ---------------------------------------------------------------------------
// Persistent integrator: each wave owns 16 rows for all steps.
// ---------------------------------------------------------------------------
__global__ __launch_bounds__(128, 1)
void heun_main_kernel(const float* __restrict__ x_in,
                      const float* __restrict__ v_in,
                      const float* __restrict__ force,
                      const float* __restrict__ bforget,
                      const unsigned int* __restrict__ wpk,
                      const int* __restrict__ steps_p,
                      float* __restrict__ out) {
  extern __shared__ float lds[];
  const int tid  = threadIdx.x;
  const int lane = tid & 31;
  const int w    = tid >> 5;
  const int r0   = blockIdx.x * 64 + w * 16;

  float* cx = lds + (size_t)w * (3 * WAVE_STATE_F);
  float* cv = cx + WAVE_STATE_F;
  float* vp = cv + WAVE_STATE_F;
  __bf16* hst = (__bf16*)(lds + HST_OFF) + (size_t)w * 512;
  float* vsqp = lds + VSQ_OFF + (size_t)w * 16;

  // stage initial state into LDS
  {
    const float4* gx = (const float4*)(x_in + (size_t)r0 * DDIM);
    const float4* gv = (const float4*)(v_in + (size_t)r0 * DDIM);
    float4* lx = (float4*)cx;
    float4* lv = (float4*)cv;
    for (int j = lane; j < 16 * (DDIM / 4); j += 32) {
      lx[j] = gx[j];
      lv[j] = gv[j];
    }
  }
  __asm__ volatile("s_wait_dscnt 0x0" ::: "memory");

  const float* force_row = force + (size_t)r0 * DDIM;
  const int nsteps = *steps_p;

  for (int s = 0; s < nsteps; ++s) {
    eval_dynamics<0>(cx, cv, cx, cv, vp, vsqp, hst, lds, wpk, force_row, bforget, lane, tid);
    eval_dynamics<1>(cx, vp, cx, cv, vp, vsqp, hst, lds, wpk, force_row, bforget, lane, tid);
  }

  // write fx then fv (concatenated outputs)
  {
    float4* gx = (float4*)(out + (size_t)r0 * DDIM);
    float4* gv = (float4*)(out + (size_t)B_ROWS * DDIM + (size_t)r0 * DDIM);
    const float4* lx = (const float4*)cx;
    const float4* lv = (const float4*)cv;
    for (int j = lane; j < 16 * (DDIM / 4); j += 32) {
      gx[j] = lx[j];
      gv[j] = lv[j];
    }
  }
}

// ---------------------------------------------------------------------------
// Host launcher
// ---------------------------------------------------------------------------
extern "C" void kernel_launch(void* const* d_in, const int* in_sizes, int n_in,
                              void* d_out, int out_size, void* d_ws, size_t ws_size,
                              hipStream_t stream) {
  (void)in_sizes; (void)n_in; (void)out_size; (void)ws_size;
  const float* x_in  = (const float*)d_in[0];
  const float* v_in  = (const float*)d_in[1];
  const float* force = (const float*)d_in[2];
  const float* U     = (const float*)d_in[3];
  const float* W     = (const float*)d_in[4];
  const float* Wf    = (const float*)d_in[5];
  const float* bfg   = (const float*)d_in[6];
  const float* Wi    = (const float*)d_in[7];
  const int*   steps = (const int*)d_in[8];

  unsigned int* wpk = (unsigned int*)d_ws;
  float* out = (float*)d_out;

  prep_weights_kernel<<<(TOTAL_PK + 255) / 256, 256, 0, stream>>>(U, W, Wf, Wi, wpk);

  const size_t shmem = (size_t)LDS_TOTAL_F * sizeof(float);  // ~260 KB per WGP
  heun_main_kernel<<<B_ROWS / 64, 128, shmem, stream>>>(
      x_in, v_in, force, bfg, wpk, steps, out);
}
